// MultiBoxLoss_80384607912707
// MI455X (gfx1250) — compile-verified, matrix operand order checked by reading, not run on hardware
//
#include <hip/hip_runtime.h>
#include <hip/hip_bf16.h>
#include <math.h>

// Problem constants (from reference setup_inputs)
#define NB  64
#define PP  24564
#define MM  16
#define CC  21
#define NPP (NB * PP)   // 1,572,096 rows; divisible by 16 (98,256 groups)

typedef __attribute__((ext_vector_type(16))) _Float16 v16h;
typedef __attribute__((ext_vector_type(8)))  float    v8f;

// ---------------------------------------------------------------------------
// 0) Re-initialize workspace state every call (d_ws is poisoned, not re-zeroed)
// ---------------------------------------------------------------------------
__global__ void init_ws_kernel(unsigned long long* __restrict__ bp_key,
                               int* __restrict__ num_pos,
                               float* __restrict__ accum) {
    const int t = threadIdx.x;
    if (t < NB * MM) bp_key[t] = 0ull;
    if (t < NB)      num_pos[t] = 0;
    if (t < 2)       accum[t]   = 0.f;
}

// ---------------------------------------------------------------------------
// 1) Matching: per (n,p) IoU against 16 truths. best_truth (per prior) kept in
//    registers; best_prior (per truth) via packed u64 argmax: LDS DS_MAX_U64
//    then one global atomicMax per block. key = (bits(ov)<<32) | (~p) so
//    larger overlap wins, ties prefer the smaller prior index (jnp.argmax).
// ---------------------------------------------------------------------------
__global__ void match_kernel(const float* __restrict__ priors,
                             const float* __restrict__ targets,
                             float* __restrict__ best_ov,
                             int*   __restrict__ best_idx,
                             unsigned long long* __restrict__ bp_key) {
    const int n = blockIdx.y;
    const int p = blockIdx.x * blockDim.x + threadIdx.x;

    __shared__ float4 s_tb[MM];
    __shared__ float  s_ta[MM];
    __shared__ unsigned long long s_key[MM];
    if (threadIdx.x < MM) {
        const float* tp = targets + (size_t)(n * MM + threadIdx.x) * 5;
        float4 b; b.x = tp[0]; b.y = tp[1]; b.z = tp[2]; b.w = tp[3];
        s_tb[threadIdx.x] = b;
        s_ta[threadIdx.x] = (b.z - b.x) * (b.w - b.y);
        s_key[threadIdx.x] = 0ull;
    }
    __syncthreads();

    if (p < PP) {
        const float4 pr = ((const float4*)priors)[p];           // (cx,cy,w,h)
        const float px0 = pr.x - pr.z * 0.5f, py0 = pr.y - pr.w * 0.5f;
        const float px1 = pr.x + pr.z * 0.5f, py1 = pr.y + pr.w * 0.5f;
        const float pa  = pr.z * pr.w;
        float bov = -1.f; int bidx = 0;                          // first-index ties
        #pragma unroll
        for (int j = 0; j < MM; ++j) {
            const float4 tb = s_tb[j];
            float iw = fminf(tb.z, px1) - fmaxf(tb.x, px0);
            float ih = fminf(tb.w, py1) - fmaxf(tb.y, py0);
            iw = fmaxf(iw, 0.f); ih = fmaxf(ih, 0.f);
            const float inter = iw * ih;
            const float ov = inter / (s_ta[j] + pa - inter);
            if (ov > bov) { bov = ov; bidx = j; }
            const unsigned long long key =
                ((unsigned long long)__float_as_uint(ov) << 32) |
                (unsigned long long)(0xFFFFFFFFu - (unsigned)p);
            atomicMax(&s_key[j], key);
        }
        const size_t row = (size_t)n * PP + p;
        best_ov[row]  = bov;
        best_idx[row] = bidx;
    }
    __syncthreads();
    if (threadIdx.x < MM)
        atomicMax(&bp_key[n * MM + threadIdx.x], s_key[threadIdx.x]);
}

// ---------------------------------------------------------------------------
// 2) Override: for each truth j (sequential -> later j wins duplicates, like
//    the scatter in the reference), force its best prior to ov=2.0, idx=j.
// ---------------------------------------------------------------------------
__global__ void override_kernel(const unsigned long long* __restrict__ bp_key,
                                float* __restrict__ best_ov,
                                int*   __restrict__ best_idx) {
    const int n = threadIdx.x;
    if (n >= NB) return;
    for (int j = 0; j < MM; ++j) {
        const unsigned p =
            0xFFFFFFFFu - (unsigned)(bp_key[n * MM + j] & 0xFFFFFFFFull);
        const size_t row = (size_t)n * PP + p;
        best_ov[row]  = 2.0f;
        best_idx[row] = j;
    }
}

// ---------------------------------------------------------------------------
// 3) conf_t + encode + masked smooth-L1. One float atomic + one int atomic
//    per block after an LDS tree reduction.
// ---------------------------------------------------------------------------
__global__ void loc_loss_kernel(const float* __restrict__ loc_data,
                                const float* __restrict__ priors,
                                const float* __restrict__ targets,
                                const float* __restrict__ best_ov,
                                const int*   __restrict__ best_idx,
                                int*   __restrict__ conf_t,
                                int*   __restrict__ num_pos,
                                float* __restrict__ accum) {
    const int n = blockIdx.y;
    const int p = blockIdx.x * blockDim.x + threadIdx.x;
    float ll = 0.f; int cnt = 0;
    if (p < PP) {
        const size_t row = (size_t)n * PP + p;
        const float ov  = best_ov[row];
        const int   idx = best_idx[row];
        const float* tp = targets + (size_t)(n * MM + idx) * 5;
        const int ct = (ov < 0.5f) ? 0 : ((int)tp[4] + 1);
        conf_t[row] = ct;
        if (ct > 0) {
            cnt = 1;
            const float4 pr = ((const float4*)priors)[p];
            const float mx0 = tp[0], my0 = tp[1], mx1 = tp[2], my1 = tp[3];
            const float gx = ((mx0 + mx1) * 0.5f - pr.x) / (0.1f * pr.z);
            const float gy = ((my0 + my1) * 0.5f - pr.y) / (0.1f * pr.w);
            const float gw = __logf((mx1 - mx0) / pr.z) * 5.0f;  // /0.2
            const float gh = __logf((my1 - my0) / pr.w) * 5.0f;
            const float* lp = loc_data + row * 4;
            const float d0 = __builtin_nontemporal_load(lp + 0) - gx;
            const float d1 = __builtin_nontemporal_load(lp + 1) - gy;
            const float d2 = __builtin_nontemporal_load(lp + 2) - gw;
            const float d3 = __builtin_nontemporal_load(lp + 3) - gh;
            #define SL1(d) (fabsf(d) < 1.f ? 0.5f * (d) * (d) : fabsf(d) - 0.5f)
            ll = SL1(d0) + SL1(d1) + SL1(d2) + SL1(d3);
            #undef SL1
        }
    }
    __shared__ float sl[256];
    __shared__ int   sc[256];
    sl[threadIdx.x] = ll; sc[threadIdx.x] = cnt;
    __syncthreads();
    for (int s = 128; s > 0; s >>= 1) {
        if (threadIdx.x < s) {
            sl[threadIdx.x] += sl[threadIdx.x + s];
            sc[threadIdx.x] += sc[threadIdx.x + s];
        }
        __syncthreads();
    }
    if (threadIdx.x == 0) {
        if (sl[0] != 0.f) atomicAdd(&accum[0], sl[0]);
        if (sc[0] != 0)   atomicAdd(&num_pos[n], sc[0]);
    }
}

// ---------------------------------------------------------------------------
// 4) Conf loss: memory-bound pass over conf_data (132 MB, streamed once ->
//    non-temporal loads so the one-shot stream doesn't evict mined[]/conf_t[]
//    from L2). Each wave handles 16 rows per iteration; Σexp over the 21
//    classes is ONE v_wmma_f32_16x16x32_f16 with an all-ones B matrix:
//      A = exp(conf - rowmax) in f16, classes 21..31 padded with 0,
//      D[m][*] = f32 row sums (replicated across N lanes).
//    ISA A-layout (16-bit, 16x32): lanes 0-15 hold K {0..7,16..23},
//    lanes 16-31 hold K {8..15,24..31}; D: VGPR j = M=j (lo half) / M=8+j.
//    The lane-varying extraction of d[j] is written as a cndmask select
//    chain (NOT if/else) to avoid EXEC-mask branch trees in the hot loop.
// ---------------------------------------------------------------------------
__global__ void conf_loss_kernel(const float* __restrict__ conf,
                                 const int*   __restrict__ conf_t,
                                 float* __restrict__ mined,
                                 float* __restrict__ accum) {
    const int lane = threadIdx.x & 31;
    const int hf   = lane >> 4;      // which half of the wave
    const int r16  = lane & 15;      // row within the 16-row group (phase 1)
    const int wave    = blockIdx.x * (blockDim.x >> 5) + (threadIdx.x >> 5);
    const int nwaves  = gridDim.x * (blockDim.x >> 5);
    const int ngroups = NPP / 16;

    v16h ones;
    #pragma unroll
    for (int i = 0; i < 16; ++i) ones[i] = (_Float16)1.0f;

    float pos_local = 0.f;

    for (int g = wave; g < ngroups; g += nwaves) {
        const int row_base = g * 16;
        const int row = row_base + r16;
        const float* rowp = conf + (size_t)row * CC;

        // Load this lane's A-slice of the row (13 classes for lo half,
        // 8 for hi half; padded K slots get -inf so exp -> 0). NT hint:
        // conf_data is streamed exactly once.
        float c[13];
        #pragma unroll
        for (int i = 0; i < 8; ++i)
            c[i] = __builtin_nontemporal_load(rowp + (hf ? (8 + i) : i));
        #pragma unroll
        for (int i = 0; i < 5; ++i)
            c[8 + i] = hf ? -INFINITY
                          : __builtin_nontemporal_load(rowp + 16 + i);

        float mloc = c[0];
        #pragma unroll
        for (int i = 1; i < 13; ++i) mloc = fmaxf(mloc, c[i]);
        const float m = fmaxf(mloc, __shfl_xor(mloc, 16, 32)); // row max

        v16h a;
        #pragma unroll
        for (int i = 0; i < 13; ++i) a[i] = (_Float16)__expf(c[i] - m);
        a[13] = (_Float16)0.f; a[14] = (_Float16)0.f; a[15] = (_Float16)0.f;

        v8f cz = {0.f, 0.f, 0.f, 0.f, 0.f, 0.f, 0.f, 0.f};
        // D = A x ones + 0 : all 16 row-sums, f32 accumulate
        v8f d = __builtin_amdgcn_wmma_f32_16x16x32_f16(
            false, a, false, ones, (short)0, cz, false, false);

        // Lanes 0-15 hold rows 0..7 in d[0..7]; lanes 16-31 hold rows 8..15.
        // Branchless lane-varying element select -> v_cndmask chain.
        const int j = r16 & 7;
        float s = d[0];
        #pragma unroll
        for (int i = 1; i < 8; ++i) s = (j == i) ? d[i] : s;
        // row max for the row this lane stores lives in lane (hf*8 + j)
        const float msel = __shfl(m, hf * 8 + j, 32);

        if (r16 < 8) {
            const int   rs  = row_base + hf * 8 + j;
            const float lse = msel + __logf(s);
            const int   ct  = conf_t[rs];
            const float gathered = conf[(size_t)rs * CC + ct];
            const float lc = lse - gathered;
            if (ct > 0) { pos_local += lc; mined[rs] = 0.f; }
            else        { mined[rs] = lc; }
        }
    }

    __shared__ float sp[256];
    sp[threadIdx.x] = pos_local;
    __syncthreads();
    for (int s = 128; s > 0; s >>= 1) {
        if (threadIdx.x < s) sp[threadIdx.x] += sp[threadIdx.x + s];
        __syncthreads();
    }
    if (threadIdx.x == 0 && sp[0] != 0.f) atomicAdd(&accum[1], sp[0]);
}

// ---------------------------------------------------------------------------
// 5) Hard-negative mining: per row, radix-select the k-th largest mined loss
//    (4 passes of 8-bit digits; mined >= 0 so float bits are monotonic),
//    then sum values > T plus (k - cnt_gt) copies of T. 6.3 MB -> all in L2.
// ---------------------------------------------------------------------------
__global__ void mine_kernel(const float* __restrict__ mined,
                            const int*   __restrict__ num_pos,
                            float* __restrict__ accum) {
    const int n = blockIdx.x;
    const float* v = mined + (size_t)n * PP;
    const int np = num_pos[n];
    const int k  = (3 * np < PP - 1) ? 3 * np : (PP - 1);

    __shared__ unsigned s_hist[256];
    __shared__ unsigned s_prefix;
    __shared__ int      s_rem;
    __shared__ float    ssum[256];
    __shared__ unsigned scnt[256];

    if (threadIdx.x == 0) { s_prefix = 0u; s_rem = k; }
    __syncthreads();
    if (k <= 0) return;

    for (int pass = 3; pass >= 0; --pass) {
        const int shift = pass * 8;
        const unsigned himask =
            (pass == 3) ? 0u : (0xFFFFFFFFu << (shift + 8));
        s_hist[threadIdx.x] = 0u;
        __syncthreads();
        const unsigned prefix = s_prefix;
        for (int i = threadIdx.x; i < PP; i += blockDim.x) {
            const unsigned u = __float_as_uint(v[i]);
            if ((u & himask) == prefix)
                atomicAdd(&s_hist[(u >> shift) & 0xFFu], 1u);
        }
        __syncthreads();
        if (threadIdx.x == 0) {
            unsigned cum = 0, bin = 0;
            const int rem = s_rem;
            for (int b = 255; b >= 0; --b) {
                const unsigned cb = s_hist[b];
                if (cum + cb >= (unsigned)rem) { bin = (unsigned)b; break; }
                cum += cb;
            }
            s_rem    = rem - (int)cum;
            s_prefix = prefix | (bin << shift);
        }
        __syncthreads();
    }

    const unsigned T  = s_prefix;           // bits of the k-th largest value
    const float    Tf = __uint_as_float(T);
    float lsum = 0.f; unsigned lcnt = 0;
    for (int i = threadIdx.x; i < PP; i += blockDim.x) {
        const float x = v[i];
        if (__float_as_uint(x) > T) { lsum += x; lcnt++; }
    }
    ssum[threadIdx.x] = lsum; scnt[threadIdx.x] = lcnt;
    __syncthreads();
    for (int s = 128; s > 0; s >>= 1) {
        if (threadIdx.x < s) {
            ssum[threadIdx.x] += ssum[threadIdx.x + s];
            scnt[threadIdx.x] += scnt[threadIdx.x + s];
        }
        __syncthreads();
    }
    if (threadIdx.x == 0) {
        const float neg = ssum[0] + (float)(k - (int)scnt[0]) * Tf;
        atomicAdd(&accum[1], neg);
    }
}

// ---------------------------------------------------------------------------
// 6) Finalize: n = total positives; out = (loss_l/n, loss_c/n)
// ---------------------------------------------------------------------------
__global__ void finalize_kernel(const int* __restrict__ num_pos,
                                const float* __restrict__ accum,
                                float* __restrict__ out) {
    __shared__ int sc[64];
    sc[threadIdx.x] = num_pos[threadIdx.x];
    __syncthreads();
    for (int s = 32; s > 0; s >>= 1) {
        if (threadIdx.x < s) sc[threadIdx.x] += sc[threadIdx.x + s];
        __syncthreads();
    }
    if (threadIdx.x == 0) {
        const float nf = (float)sc[0];
        out[0] = accum[0] / nf;
        out[1] = accum[1] / nf;
    }
}

// ---------------------------------------------------------------------------
extern "C" void kernel_launch(void* const* d_in, const int* in_sizes, int n_in,
                              void* d_out, int out_size, void* d_ws, size_t ws_size,
                              hipStream_t stream) {
    (void)in_sizes; (void)n_in; (void)out_size; (void)ws_size;
    const float* loc_data  = (const float*)d_in[0];
    const float* conf_data = (const float*)d_in[1];
    const float* priors    = (const float*)d_in[2];
    const float* targets   = (const float*)d_in[3];
    float* out = (float*)d_out;

    // Workspace layout (~25.2 MB):
    float* best_ov  = (float*)d_ws;                                // NPP f32
    int*   best_idx = (int*)(best_ov + NPP);                       // NPP i32
    int*   conf_t   = (int*)(best_idx + NPP);                      // NPP i32
    float* mined    = (float*)(conf_t + NPP);                      // NPP f32
    unsigned long long* bp_key =
        (unsigned long long*)(mined + NPP);                        // NB*MM u64
    int*   num_pos  = (int*)(bp_key + NB * MM);                    // NB i32
    float* accum    = (float*)(num_pos + NB);                      // 2 f32

    init_ws_kernel<<<1, 1024, 0, stream>>>(bp_key, num_pos, accum);

    dim3 gridA((PP + 255) / 256, NB);
    match_kernel<<<gridA, 256, 0, stream>>>(priors, targets,
                                            best_ov, best_idx, bp_key);
    override_kernel<<<1, 64, 0, stream>>>(bp_key, best_ov, best_idx);
    loc_loss_kernel<<<gridA, 256, 0, stream>>>(loc_data, priors, targets,
                                               best_ov, best_idx,
                                               conf_t, num_pos, accum);
    conf_loss_kernel<<<1024, 256, 0, stream>>>(conf_data, conf_t, mined, accum);
    mine_kernel<<<NB, 256, 0, stream>>>(mined, num_pos, accum);
    finalize_kernel<<<1, 64, 0, stream>>>(num_pos, accum, out);
}
